// ContinuousValueEncoder_41171556500000
// MI455X (gfx1250) — matrix operand order repaired and checked by reading, not compiled, not run on hardware
//
#include <hip/hip_runtime.h>

typedef __attribute__((ext_vector_type(16))) _Float16 v16h;
typedef __attribute__((ext_vector_type(8)))  _Float16 v8h;
typedef __attribute__((ext_vector_type(4)))  _Float16 v4h;
typedef __attribute__((ext_vector_type(8)))  float    v8f;

#define D_MODEL 512
#define ROWS_PB 64          // rows (M) per workgroup tile
#define HPAD    520         // 512 + 8 f16 pad -> 1040B row stride, bank-conflict free
#define KT_N    16          // 512 / 32 K-steps

// ---------------------------------------------------------------------------
// Pre-pass: W2 (512x512 f32, row-major [e][d]) -> f16 row-major in workspace.
// B-matrix fragment for wmma then = 32 contiguous bytes per lane, L2-resident.
// ---------------------------------------------------------------------------
__global__ __launch_bounds__(256) void w2_to_f16(const float* __restrict__ W2,
                                                 _Float16* __restrict__ W2h) {
  const int idx = blockIdx.x * 256 + threadIdx.x;     // 65536 threads, 4 elems each
  const float4 v = ((const float4*)W2)[idx];
  v4h h;
  h[0] = (_Float16)v.x; h[1] = (_Float16)v.y;
  h[2] = (_Float16)v.z; h[3] = (_Float16)v.w;
  ((v4h*)W2h)[idx] = h;
}

// ---------------------------------------------------------------------------
// Fused: h = relu(min(x,512)*W1+b1) (LDS, f16) -> h @ W2^T + b2 (WMMA f16->f32)
//        -> LayerNorm (register epilogue, cross-wave LDS reduce) -> mask -> out
// Block: 512 threads (16 waves). Tile: 64 rows x 512 cols.
// Wave w: cols [32w, 32w+32), M-subtiles 0..3, 8 f32 accumulators.
// ---------------------------------------------------------------------------
__global__ __launch_bounds__(512) void cve_main(
    const float* __restrict__ x,  const float* __restrict__ W1,
    const float* __restrict__ b1, const _Float16* __restrict__ W2h,
    const float* __restrict__ b2, const float* __restrict__ gamma,
    const float* __restrict__ beta, float* __restrict__ out)
{
  __shared__ float    sW1[D_MODEL];
  __shared__ float    sb1[D_MODEL];
  __shared__ _Float16 sH[ROWS_PB * HPAD];   // 66.5 KB f16 activations
  __shared__ float    sPart[16][ROWS_PB][2];
  __shared__ float    sMu[ROWS_PB];
  __shared__ float    sRs[ROWS_PB];
  __shared__ float    sVal[ROWS_PB];

  const int tid  = threadIdx.x;
  const int wave = tid >> 5;        // 0..15
  const int lane = tid & 31;
  const int l16  = lane & 15;
  const int hl   = lane >> 4;       // lane half (wave32)
  const long rowbase = (long)blockIdx.x * ROWS_PB;

  // Stage W1/b1 (2KB each) into LDS.
  sW1[tid] = W1[tid & (D_MODEL - 1)];
  sb1[tid] = b1[tid & (D_MODEL - 1)];
  __syncthreads();

  // Compute h tile into LDS as f16. Thread t: row = t&63, k-chunk = (t>>6)*64.
  {
    const int r  = tid & 63;
    const int k0 = (tid >> 6) << 6;
    const float xv = fminf(x[rowbase + r], 512.0f);
    _Float16* dst = &sH[r * HPAD + k0];
    #pragma unroll
    for (int j = 0; j < 64; j += 8) {
      v8h p;
      #pragma unroll
      for (int u = 0; u < 8; ++u)
        p[u] = (_Float16)fmaxf(fmaf(xv, sW1[k0 + j + u], sb1[k0 + j + u]), 0.0f);
      *(v8h*)(dst + j) = p;        // 16B ds_store_b128, conflict-free
    }
  }
  __syncthreads();

  // ---- WMMA main loop ----
  v8f acc[4][2];
  #pragma unroll
  for (int mt = 0; mt < 4; ++mt)
    #pragma unroll
    for (int nt = 0; nt < 2; ++nt)
      acc[mt][nt] = (v8f){0.f,0.f,0.f,0.f,0.f,0.f,0.f,0.f};

  const int n0 = wave << 5;                               // column base for this wave
  // B fragment: lane column n = n0 + nt*16 + l16 ; K = kt*32 + hl*16 + [0..15]
  const _Float16* B0 = W2h + (long)(n0 + l16) * D_MODEL + hl * 16;
  const _Float16* B1 = B0 + 16 * D_MODEL;

  for (int kt = 0; kt < KT_N; ++kt) {
    const v16h bf0 = *(const v16h*)(B0 + kt * 32);        // 32B, L2-hot
    const v16h bf1 = *(const v16h*)(B1 + kt * 32);
    // A fragment: lane row m = l16, K = kt*32 + hl*16 + [0..15]
    const _Float16* arow = &sH[l16 * HPAD + kt * 32 + hl * 16];
    #pragma unroll
    for (int mt = 0; mt < 4; ++mt) {
      const _Float16* ap = arow + mt * 16 * HPAD;
      const v8h alo = *(const v8h*)(ap);                  // ds_load_b128
      const v8h ahi = *(const v8h*)(ap + 8);              // ds_load_b128
      v16h af;
      #pragma unroll
      for (int u = 0; u < 8; ++u) { af[u] = alo[u]; af[u + 8] = ahi[u]; }
      acc[mt][0] = __builtin_amdgcn_wmma_f32_16x16x32_f16(
          false, af, false, bf0, (short)0, acc[mt][0], false, false);
      acc[mt][1] = __builtin_amdgcn_wmma_f32_16x16x32_f16(
          false, af, false, bf1, (short)0, acc[mt][1], false, false);
    }
  }

  // ---- epilogue: +b2, per-row sum / sumsq partials ----
  const float b2v0 = b2[n0 + l16];
  const float b2v1 = b2[n0 + 16 + l16];
  #pragma unroll
  for (int mt = 0; mt < 4; ++mt) {
    #pragma unroll
    for (int r = 0; r < 8; ++r) {
      const float a0 = acc[mt][0][r] + b2v0;
      const float a1 = acc[mt][1][r] + b2v1;
      acc[mt][0][r] = a0;
      acc[mt][1][r] = a1;
      float s = a0 + a1;
      float q = a0 * a0 + a1 * a1;
      s += __shfl_xor(s, 1); s += __shfl_xor(s, 2);
      s += __shfl_xor(s, 4); s += __shfl_xor(s, 8);   // reduce within 16-lane half
      q += __shfl_xor(q, 1); q += __shfl_xor(q, 2);
      q += __shfl_xor(q, 4); q += __shfl_xor(q, 8);
      if (l16 == 0) {
        const int row = mt * 16 + hl * 8 + r;         // C layout: M = r + 8*hl
        sPart[wave][row][0] = s;
        sPart[wave][row][1] = q;
      }
    }
  }
  __syncthreads();

  // Cross-wave LN stats: one thread per row.
  if (tid < ROWS_PB) {
    float s = 0.f, q = 0.f;
    #pragma unroll
    for (int w = 0; w < 16; ++w) { s += sPart[w][tid][0]; q += sPart[w][tid][1]; }
    const float mu  = s * (1.0f / D_MODEL);
    const float var = q * (1.0f / D_MODEL) - mu * mu;
    sMu[tid]  = mu;
    sRs[tid]  = rsqrtf(var + 1e-5f);
    sVal[tid] = (x[rowbase + tid] >= 0.0f) ? 1.0f : 0.0f;   // row mask
  }
  __syncthreads();

  // ---- normalize in registers and store (only HBM touch of h2) ----
  const float g0  = gamma[n0 + l16],     g1  = gamma[n0 + 16 + l16];
  const float be0 = beta[n0 + l16],      be1 = beta[n0 + 16 + l16];
  #pragma unroll
  for (int mt = 0; mt < 4; ++mt) {
    #pragma unroll
    for (int r = 0; r < 8; ++r) {
      const int row = mt * 16 + hl * 8 + r;
      const float mu = sMu[row], rs = sRs[row], vm = sVal[row];
      float* orow = out + (rowbase + row) * D_MODEL;
      orow[n0 + l16]      = ((acc[mt][0][r] - mu) * rs * g0 + be0) * vm;
      orow[n0 + 16 + l16] = ((acc[mt][1][r] - mu) * rs * g1 + be1) * vm;
    }
  }
}

// ---------------------------------------------------------------------------
extern "C" void kernel_launch(void* const* d_in, const int* in_sizes, int n_in,
                              void* d_out, int out_size, void* d_ws, size_t ws_size,
                              hipStream_t stream) {
  const float* x  = (const float*)d_in[0];
  const float* W1 = (const float*)d_in[1];
  const float* b1 = (const float*)d_in[2];
  const float* W2 = (const float*)d_in[3];
  const float* b2 = (const float*)d_in[4];
  const float* gm = (const float*)d_in[5];
  const float* bt = (const float*)d_in[6];
  float* out = (float*)d_out;
  _Float16* W2h = (_Float16*)d_ws;                 // 512 KB f16 weights

  w2_to_f16<<<(D_MODEL * D_MODEL / 4) / 256, 256, 0, stream>>>(W2, W2h);

  const int rows = in_sizes[0];                    // B*S = 65536
  cve_main<<<rows / ROWS_PB, 512, 0, stream>>>(x, W1, b1, W2h, b2, gm, bt, out);
}
